// GlimpseNetwork_38199439131201
// MI455X (gfx1250) — matrix operand order repaired and checked by reading, not compiled, not run on hardware
//
#include <hip/hip_runtime.h>
#include <hip/hip_bf16.h>

typedef __attribute__((ext_vector_type(16))) _Float16 v16h;
typedef __attribute__((ext_vector_type(8)))  float    v8f;

namespace {
constexpr int T_DIM = 8192;
constexpr int B_DIM = 4096;
constexpr int NHID  = 256;
constexpr int FANIN = 49;   // 3*16 + 1
constexpr int KPAD  = 64;   // FANIN padded to 2x K=32 WMMA steps
}

// ---------------------------------------------------------------------------
// Phase 1: per-(batch, query) binary-search linear interpolation.
// Block = one batch column (64 threads; 0..47 are the 48 glimpse queries).
// Writes g_ext [B, 64] as f16 (col 48 = l_t, cols 49..63 = 0) and the
// tuple's second output g[:, 24].
// ---------------------------------------------------------------------------
__global__ void glimpse_interp_kernel(const float* __restrict__ ts,   // [T, B] time-major
                                      const float* __restrict__ vs,   // [T, B]
                                      const float* __restrict__ lt,   // [B, 1]
                                      _Float16* __restrict__ gext,    // [B, 64]
                                      float* __restrict__ midout) {   // [B]
    const int b = blockIdx.x;
    const int q = threadIdx.x;          // 0..63
    const float l = lt[b];

    if (q >= 48) {
        gext[(size_t)b * KPAD + q] = (q == 48) ? (_Float16)l : (_Float16)0.0f;
        return;
    }

    // g flatten order is [nglimpse, ngran]: q = j*3 + i
    const int j = q / 3;                // glimpse 0..15
    const int i = q - j * 3;            // gran    0..2
    const float w = 0.2f * (float)(i * 5 + 1);
    const float x = l + (-0.5f * w + (float)j * (w * (1.0f / 15.0f)));

    const float* t = ts + b;            // column b, stride B_DIM
    const float* v = vs + b;

    // upper_bound: first index with t[idx] > x  (13 strided loads)
    int lo = 0, hi = T_DIM;
    while (lo < hi) {
        const int mid = (lo + hi) >> 1;
        const float tm = t[(size_t)mid * B_DIM];
        if (tm <= x) lo = mid + 1; else hi = mid;
    }

    float r;
    if (lo <= 0) {
        r = v[0];                                          // left fill = v[0]
    } else if (lo >= T_DIM) {
        r = v[(size_t)(T_DIM - 1) * B_DIM];                // right fill = v[-1]
    } else {
        const float t0 = t[(size_t)(lo - 1) * B_DIM];
        const float t1 = t[(size_t)lo * B_DIM];
        const float v0 = v[(size_t)(lo - 1) * B_DIM];
        const float v1 = v[(size_t)lo * B_DIM];
        const float dt = t1 - t0;
        r = (dt > 0.0f) ? v0 + (x - t0) * (v1 - v0) / dt : v0;
    }

    gext[(size_t)b * KPAD + q] = (_Float16)r;
    if (q == 24) midout[b] = r;          // g[:, 48//2]
}

// ---------------------------------------------------------------------------
// Phase 2a: pre-pack W [256, 49] (row-major) into per-lane B fragments (contiguous k, same order as the A loads),
// padded K->64, as f16. Layout: [nTile(16)][kstep(2)][lane(32)][h(16)] where
// for lane L: n = L&15, k = kstep*32 + 16*(L>>4) + h.
// ---------------------------------------------------------------------------
__global__ void pack_w_kernel(const float* __restrict__ W,
                              _Float16* __restrict__ wpack) {
    const int nt   = blockIdx.x >> 1;        // 0..15
    const int ks   = blockIdx.x & 1;         // 0..1
    const int lane = threadIdx.x;            // 0..31
    const int col  = nt * 16 + (lane & 15);  // output-feature index n
    const int kb   = ks * 32 + 16 * (lane >> 4);
    _Float16* dst = wpack + (((size_t)blockIdx.x * 32) + lane) * 16;
#pragma unroll
    for (int h = 0; h < 16; ++h) {
        const int k = kb + h;
        const float val = (k < FANIN) ? W[(size_t)col * FANIN + k] : 0.0f;
        dst[h] = (_Float16)val;
    }
}

// ---------------------------------------------------------------------------
// Phase 2b: grep = g_ext @ W^T + bias via v_wmma_f32_16x16x32_f16.
// One wave per 16x16 output tile; 8 waves / block; 4096 tiles total.
// A-fragment (16-bit A 16x32): lane L holds row m = L&15; halves 0..15 map to
// K = kBase + 16*(L>>4) + {0..15} -> two uint4 loads.
// ---------------------------------------------------------------------------
__global__ void glimpse_gemm_kernel(const _Float16* __restrict__ gext,   // [B, 64]
                                    const _Float16* __restrict__ wpack,  // packed B frags
                                    const float* __restrict__ bias,      // [256]
                                    float* __restrict__ out) {           // [B, 256]
    const int lane = threadIdx.x & 31;
    const int wv   = threadIdx.x >> 5;
    const int tile = blockIdx.x * 8 + wv;     // 0..4095
    const int nTile = tile & 15;
    const int mTile = tile >> 4;

    const int m  = lane & 15;
    const int hi = lane >> 4;
    const size_t rowBase = (size_t)(mTile * 16 + m) * KPAD;

    v8f c = {};
#pragma unroll
    for (int ks = 0; ks < 2; ++ks) {
        union { v16h v; uint4 u[2]; } A, Bf;
        const _Float16* ap = gext + rowBase + ks * 32 + 16 * hi;
        A.u[0] = *(const uint4*)(ap);         // halves 0..7  : K = base..base+7
        A.u[1] = *(const uint4*)(ap + 8);     // halves 8..15 : K = base+8..base+15
        const _Float16* bp = wpack + (((size_t)(nTile * 2 + ks) * 32) + lane) * 16;
        Bf.u[0] = *(const uint4*)(bp);
        Bf.u[1] = *(const uint4*)(bp + 8);
        c = __builtin_amdgcn_wmma_f32_16x16x32_f16(
                /*neg_a=*/false, A.v, /*neg_b=*/false, Bf.v,
                /*c_mod=*/(short)0, c, /*reuse_a=*/false, /*reuse_b=*/false);
    }

    // D layout: c[r] -> (M = mTile*16 + r + 8*hi, N = nTile*16 + (lane&15))
    const int col = nTile * 16 + (lane & 15);
    const float bb = bias[col];
#pragma unroll
    for (int r = 0; r < 8; ++r) {
        out[(size_t)(mTile * 16 + r + 8 * hi) * NHID + col] = c[r] + bb;
    }
}

// ---------------------------------------------------------------------------
extern "C" void kernel_launch(void* const* d_in, const int* in_sizes, int n_in,
                              void* d_out, int out_size, void* d_ws, size_t ws_size,
                              hipStream_t stream) {
    const float* timesteps = (const float*)d_in[0];   // [T, B]
    const float* values    = (const float*)d_in[1];   // [T, B]
    const float* l_t       = (const float*)d_in[2];   // [B, 1]
    const float* W         = (const float*)d_in[3];   // [NHID, 49]
    const float* bias      = (const float*)d_in[4];   // [NHID]

    float* grep = (float*)d_out;                       // [B, NHID]
    float* mid  = (float*)d_out + (size_t)B_DIM * NHID;// [B]

    _Float16* gext  = (_Float16*)d_ws;                           // B*64 f16 = 512 KB
    _Float16* wpack = (_Float16*)((char*)d_ws + (size_t)B_DIM * KPAD * sizeof(_Float16)); // 32 KB

    pack_w_kernel<<<32, 32, 0, stream>>>(W, wpack);
    glimpse_interp_kernel<<<B_DIM, 64, 0, stream>>>(timesteps, values, l_t, gext, mid);
    glimpse_gemm_kernel<<<(B_DIM / 16) * (NHID / 16) / 8, 256, 0, stream>>>(gext, wpack, bias, grep);
}
